// StaticExpansionBlock_56238301774415
// MI455X (gfx1250) — compile-verified, matrix operand order checked by reading, not run on hardware
//
#include <hip/hip_runtime.h>
#include <hip/hip_bf16.h>

typedef __attribute__((ext_vector_type(16))) __bf16 v16bf;
typedef __attribute__((ext_vector_type(8)))  __bf16 v8bf;
typedef __attribute__((ext_vector_type(8)))  float  v8f;

#define BS 32
#define LQ 512
#define DM 1024
#define SS 992
#define SP 1024
#define EPSV 1e-9f

__device__ __forceinline__ __bf16 to_bf16(float f) {
    unsigned u = __float_as_uint(f);
    unsigned r = (u + 0x7FFFu + ((u >> 16) & 1u)) >> 16;
    unsigned short us = (unsigned short)r;
    return __builtin_bit_cast(__bf16, us);
}

// Load a 16x32 bf16 WMMA A/B fragment from row-major [rows, ld] storage.
// p points at element [lane%16, 0] of the fragment (caller pre-adds (lane&15)*ld
// as part of the offset). koff = k + (lane>=16 ? 8 : 0). Per the CDNA5 ISA bf16
// 16x32 layout, each lane holds two contiguous 8-half runs: k in [klo, klo+8)
// and [klo+16, klo+24).
__device__ __forceinline__ v16bf load_frag(const __bf16* base, size_t off, int koff) {
    const __bf16* p = base + off + koff;
    union { v16bf f; v8bf h[2]; } u;
    u.h[0] = *reinterpret_cast<const v8bf*>(p);
    u.h[1] = *reinterpret_cast<const v8bf*>(p + 16);
    return u.f;
}

__device__ __forceinline__ void store_frag(float* C, int ldc, int mbase, int col,
                                           const v8f& a, const float* bias, float scale) {
    float bv = bias ? bias[col] : 0.0f;
#pragma unroll
    for (int i = 0; i < 8; ++i)
        C[(size_t)(mbase + i) * ldc + col] = a[i] * scale + bv;
}

// Batched C[m,n] = scale * sum_k A[m,k]*B[n,k] + bias[n]
// A: [M,K] bf16 (optionally row-gathered via rowIdx), B: [N,K] bf16, C: [M,N] f32.
// Block: 256 threads = 8 waves as 2(m) x 4(n); block tile 128x256; wave tile 64x64
// (4x4 accumulators of 16x16). Requires M%128==0, N%256==0, K%32==0.
__global__ __launch_bounds__(256)
void wmma_gemm_bf16(const __bf16* __restrict__ A, const __bf16* __restrict__ B,
                    const float* __restrict__ bias, float* __restrict__ C,
                    int K, int lda, int ldb, int ldc,
                    long long strideA, long long strideB, long long strideC,
                    const int* __restrict__ rowIdx, int idxStride, int idxCount,
                    float scale) {
    const int lane = threadIdx.x & 31;
    const int wave = threadIdx.x >> 5;
    const int wm = wave >> 2;      // 0..1
    const int wn = wave & 3;       // 0..3
    const int b  = blockIdx.z;
    const int m0 = blockIdx.y * 128 + wm * 64;
    const int n0 = blockIdx.x * 256 + wn * 64;
    const int r  = lane & 15;
    const int hi = lane >> 4;
    const int koff0 = hi * 8;

    const __bf16* Ab = A + (size_t)b * strideA;
    const __bf16* Bb = B + (size_t)b * strideB;

    // Integer row offsets (keeps global address space through the gather phi).
    size_t aoff[4];
#pragma unroll
    for (int i = 0; i < 4; ++i) {
        int m = m0 + i * 16 + r;
        int gi = m;
        if (rowIdx) {
            const int* ix = rowIdx + (size_t)b * idxStride;
            gi = (m < idxCount) ? ix[m] : 0;
        }
        aoff[i] = (size_t)gi * lda;
    }
    size_t boff[4];
#pragma unroll
    for (int j = 0; j < 4; ++j)
        boff[j] = (size_t)(n0 + j * 16 + r) * ldb;

    v8f acc[4][4];
#pragma unroll
    for (int i = 0; i < 4; ++i)
#pragma unroll
        for (int j = 0; j < 4; ++j)
            acc[i][j] = (v8f){};

    for (int k = 0; k < K; k += 32) {
        const int ko = k + koff0;
        // Load order: operands of the first WMMA group arrive first.
        v16bf a0 = load_frag(Ab, aoff[0], ko);
        v16bf b0 = load_frag(Bb, boff[0], ko);
        v16bf b1 = load_frag(Bb, boff[1], ko);
        v16bf b2 = load_frag(Bb, boff[2], ko);
        v16bf b3 = load_frag(Bb, boff[3], ko);
        v16bf a1 = load_frag(Ab, aoff[1], ko);
        v16bf a2 = load_frag(Ab, aoff[2], ko);
        v16bf a3 = load_frag(Ab, aoff[3], ko);

        acc[0][0] = __builtin_amdgcn_wmma_f32_16x16x32_bf16(false, a0, false, b0, (short)0, acc[0][0], false, false);
        acc[0][1] = __builtin_amdgcn_wmma_f32_16x16x32_bf16(false, a0, false, b1, (short)0, acc[0][1], false, false);
        acc[0][2] = __builtin_amdgcn_wmma_f32_16x16x32_bf16(false, a0, false, b2, (short)0, acc[0][2], false, false);
        acc[0][3] = __builtin_amdgcn_wmma_f32_16x16x32_bf16(false, a0, false, b3, (short)0, acc[0][3], false, false);
        acc[1][0] = __builtin_amdgcn_wmma_f32_16x16x32_bf16(false, a1, false, b0, (short)0, acc[1][0], false, false);
        acc[1][1] = __builtin_amdgcn_wmma_f32_16x16x32_bf16(false, a1, false, b1, (short)0, acc[1][1], false, false);
        acc[1][2] = __builtin_amdgcn_wmma_f32_16x16x32_bf16(false, a1, false, b2, (short)0, acc[1][2], false, false);
        acc[1][3] = __builtin_amdgcn_wmma_f32_16x16x32_bf16(false, a1, false, b3, (short)0, acc[1][3], false, false);
        acc[2][0] = __builtin_amdgcn_wmma_f32_16x16x32_bf16(false, a2, false, b0, (short)0, acc[2][0], false, false);
        acc[2][1] = __builtin_amdgcn_wmma_f32_16x16x32_bf16(false, a2, false, b1, (short)0, acc[2][1], false, false);
        acc[2][2] = __builtin_amdgcn_wmma_f32_16x16x32_bf16(false, a2, false, b2, (short)0, acc[2][2], false, false);
        acc[2][3] = __builtin_amdgcn_wmma_f32_16x16x32_bf16(false, a2, false, b3, (short)0, acc[2][3], false, false);
        acc[3][0] = __builtin_amdgcn_wmma_f32_16x16x32_bf16(false, a3, false, b0, (short)0, acc[3][0], false, false);
        acc[3][1] = __builtin_amdgcn_wmma_f32_16x16x32_bf16(false, a3, false, b1, (short)0, acc[3][1], false, false);
        acc[3][2] = __builtin_amdgcn_wmma_f32_16x16x32_bf16(false, a3, false, b2, (short)0, acc[3][2], false, false);
        acc[3][3] = __builtin_amdgcn_wmma_f32_16x16x32_bf16(false, a3, false, b3, (short)0, acc[3][3], false, false);
    }

    float* Cb = C + (size_t)b * strideC;
#pragma unroll
    for (int i = 0; i < 4; ++i) {
        const int mlo = m0 + i * 16 + hi * 8;
#pragma unroll
        for (int j = 0; j < 4; ++j)
            store_frag(Cb, ldc, mlo, n0 + j * 16 + r, acc[i][j], bias, scale);
    }
}

__global__ __launch_bounds__(256)
void cvt_f32_bf16(const float* __restrict__ in, __bf16* __restrict__ out, size_t n) {
    size_t i = (size_t)blockIdx.x * blockDim.x + threadIdx.x;
    if (i < n) out[i] = to_bf16(in[i]);
}

// outT[b,d,l] = bf16( g[b,l,d] * sigmoid(s[b,l,d]) )   (32x32 LDS transpose tiles)
__global__ __launch_bounds__(256)
void gate_transpose(const float* __restrict__ g, const float* __restrict__ s,
                    __bf16* __restrict__ outT) {
    __shared__ float tile[32][33];
    const int b = blockIdx.z;
    const int d0 = blockIdx.x * 32;
    const int l0 = blockIdx.y * 32;
    const int tx = threadIdx.x & 31;
    const int ty = threadIdx.x >> 5;
    const float* gb = g + (size_t)b * LQ * DM;
    const float* sb = s + (size_t)b * LQ * DM;
    for (int i = ty; i < 32; i += 8) {
        size_t idx = (size_t)(l0 + i) * DM + (d0 + tx);
        float sv = sb[idx];
        tile[i][tx] = gb[idx] * (1.0f / (1.0f + __expf(-sv)));
    }
    __syncthreads();
    __bf16* ob = outT + (size_t)b * DM * LQ;
    for (int i = ty; i < 32; i += 8)
        ob[(size_t)(d0 + i) * LQ + (l0 + tx)] = to_bf16(tile[tx][i]);
}

// Forward norm: per (b,s) row of z[b, SP, L]: masked relu(+-z), row-normalize, bf16 out.
// Pad rows s>=SS write zeros.
__global__ __launch_bounds__(256)
void fw_norm(const float* __restrict__ z, const int* __restrict__ mask,
             __bf16* __restrict__ fwa, __bf16* __restrict__ fwb) {
    const int s = blockIdx.x;   // 0..SP-1
    const int b = blockIdx.y;
    const int t = threadIdx.x;
    __shared__ float sa[256], sb[256];
    const float* zr = z + ((size_t)b * SP + s) * LQ;
    float va[2], vb[2];
    float pa = 0.0f, pb = 0.0f;
    if (s < SS) {
#pragma unroll
        for (int j = 0; j < 2; ++j) {
            int l = t + j * 256;
            float v = zr[l];
            int m = mask[(size_t)b * LQ + l];
            float ra = m ? fmaxf(v, 0.0f) : 0.0f;
            float rb = m ? fmaxf(-v, 0.0f) : 0.0f;
            va[j] = ra; vb[j] = rb; pa += ra; pb += rb;
        }
    } else {
        va[0] = va[1] = vb[0] = vb[1] = 0.0f;
    }
    sa[t] = pa; sb[t] = pb;
    __syncthreads();
    for (int off = 128; off > 0; off >>= 1) {
        if (t < off) { sa[t] += sa[t + off]; sb[t] += sb[t + off]; }
        __syncthreads();
    }
    float ia = 1.0f / (sa[0] + EPSV);
    float ib = 1.0f / (sb[0] + EPSV);
    __bf16* fa = fwa + ((size_t)b * SP + s) * LQ;
    __bf16* fb = fwb + ((size_t)b * SP + s) * LQ;
#pragma unroll
    for (int j = 0; j < 2; ++j) {
        int l = t + j * 256;
        fa[l] = to_bf16(va[j] * ia);
        fb[l] = to_bf16(vb[j] * ib);
    }
}

// Backward (segment) norm: per (b,l) column of z: relu(+-z) renormalized within
// segments {32,64,128,256,512}; bf16 out rows [b,l, s<SS] with ld=SS.
__global__ __launch_bounds__(256)
void bw_norm(const float* __restrict__ z, __bf16* __restrict__ bwa, __bf16* __restrict__ bwb) {
    const int l = blockIdx.x;
    const int b = blockIdx.y;
    const int t = threadIdx.x;
    __shared__ float col[SS];
    __shared__ float red[256];
    __shared__ float sumA[5], sumB[5];
    const float* zb = z + (size_t)b * SP * LQ + l;
    for (int s = t; s < SS; s += 256) col[s] = zb[(size_t)s * LQ];
    __syncthreads();
    const int seg[6] = {0, 32, 96, 224, 480, 992};
    for (int g = 0; g < 5; ++g) {
        float pa = 0.0f, pb = 0.0f;
        for (int s = seg[g] + t; s < seg[g + 1]; s += 256) {
            float v = col[s];
            pa += fmaxf(v, 0.0f);
            pb += fmaxf(-v, 0.0f);
        }
        red[t] = pa; __syncthreads();
        for (int off = 128; off > 0; off >>= 1) { if (t < off) red[t] += red[t + off]; __syncthreads(); }
        if (t == 0) sumA[g] = red[0];
        __syncthreads();
        red[t] = pb; __syncthreads();
        for (int off = 128; off > 0; off >>= 1) { if (t < off) red[t] += red[t + off]; __syncthreads(); }
        if (t == 0) sumB[g] = red[0];
        __syncthreads();
    }
    __bf16* oa = bwa + ((size_t)b * LQ + l) * SS;
    __bf16* ob = bwb + ((size_t)b * LQ + l) * SS;
    for (int s = t; s < SS; s += 256) {
        int g = (s < 32) ? 0 : (s < 96) ? 1 : (s < 224) ? 2 : (s < 480) ? 3 : 4;
        float v = col[s];
        oa[s] = to_bf16(fmaxf(v, 0.0f)  / (sumA[g] + EPSV));
        ob[s] = to_bf16(fmaxf(-v, 0.0f) / (sumB[g] + EPSV));
    }
}

// outT[b,d,s] = bf16( cls[b,s,d] + bias_table[nidx[b,s], d] ), s < SS, ld_s = SS.
__global__ __launch_bounds__(256)
void bias_transpose(const float* __restrict__ cls, const float* __restrict__ btab,
                    const int* __restrict__ nidx, __bf16* __restrict__ outT) {
    __shared__ float tile[32][33];
    const int b = blockIdx.z;
    const int s0 = blockIdx.x * 32;   // 31 tiles -> exactly SS
    const int d0 = blockIdx.y * 32;
    const int tx = threadIdx.x & 31;
    const int ty = threadIdx.x >> 5;
    const float* cb = cls + (size_t)b * SP * DM;
    const int* ix = nidx + (size_t)b * SS;
    for (int i = ty; i < 32; i += 8) {
        int s = s0 + i, d = d0 + tx;
        int row = ix[s];
        tile[i][tx] = cb[(size_t)s * DM + d] + btab[(size_t)row * DM + d];
    }
    __syncthreads();
    __bf16* ob = outT + (size_t)b * DM * SS;
    for (int i = ty; i < 32; i += 8)
        ob[(size_t)(d0 + i) * SS + (s0 + tx)] = to_bf16(tile[tx][i]);
}

// out = 0.2 * ( sigmoid(sel)*oa + (1-sigmoid(sel))*ob )
__global__ __launch_bounds__(256)
void final_combine(const float* __restrict__ sel, const float* __restrict__ oa,
                   const float* __restrict__ ob, float* __restrict__ out, size_t n) {
    size_t i = (size_t)blockIdx.x * blockDim.x + threadIdx.x;
    if (i < n) {
        float sg = 1.0f / (1.0f + __expf(-sel[i]));
        out[i] = 0.2f * (sg * oa[i] + (1.0f - sg) * ob[i]);
    }
}

static void launch_gemm(const __bf16* A, const __bf16* B, const float* bias, float* C,
                        int M, int N, int K, int lda, int ldb, int ldc,
                        long long sA, long long sB, long long sC, int batches,
                        const int* rowIdx, int idxStride, int idxCount,
                        float scale, hipStream_t st) {
    dim3 g(N / 256, M / 128, batches), blk(256);
    hipLaunchKernelGGL(wmma_gemm_bf16, g, blk, 0, st,
                       A, B, bias, C, K, lda, ldb, ldc, sA, sB, sC,
                       rowIdx, idxStride, idxCount, scale);
}

static void launch_cvt(const float* in, __bf16* out, size_t n, hipStream_t st) {
    dim3 g((unsigned)((n + 255) / 256)), blk(256);
    hipLaunchKernelGGL(cvt_f32_bf16, g, blk, 0, st, in, out, n);
}

extern "C" void kernel_launch(void* const* d_in, const int* in_sizes, int n_in,
                              void* d_out, int out_size, void* d_ws, size_t ws_size,
                              hipStream_t stream) {
    const float* x    = (const float*)d_in[0];
    const int*   nidx = (const int*)  d_in[1];
    const int*   mask = (const int*)  d_in[2];
    const float* qtab = (const float*)d_in[3];
    const float* btab = (const float*)d_in[4];
    const float* Wk  = (const float*)d_in[5];  const float* bk   = (const float*)d_in[6];
    const float* Wa  = (const float*)d_in[7];  const float* ba   = (const float*)d_in[8];
    const float* Wa1 = (const float*)d_in[9];  const float* ba1  = (const float*)d_in[10];
    const float* Wb  = (const float*)d_in[11]; const float* bb   = (const float*)d_in[12];
    const float* Wb1 = (const float*)d_in[13]; const float* bb1  = (const float*)d_in[14];
    const float* Wsl = (const float*)d_in[15]; const float* bsel = (const float*)d_in[16];
    float* out = (float*)d_out;

    const size_t NX = (size_t)BS * LQ * DM;        // 16,777,216
    const size_t NW = (size_t)DM * DM;             // 1,048,576

    char* ws = (char*)d_ws;
    size_t cur = 0;
    auto carve = [&](size_t bytes) -> char* {
        char* p = ws + cur;
        cur += (bytes + 255) & ~(size_t)255;
        return p;
    };

    __bf16* XB   = (__bf16*)carve(NX * 2);                    // bf16(x)  [B*L, D]
    __bf16* WKB  = (__bf16*)carve(NW * 2);
    __bf16* WAB  = (__bf16*)carve(NW * 2);
    __bf16* WA1B = (__bf16*)carve(NW * 2);
    __bf16* WBB  = (__bf16*)carve(NW * 2);
    __bf16* WB1B = (__bf16*)carve(NW * 2);
    __bf16* WSB  = (__bf16*)carve(NW * 2);
    __bf16* QTB  = (__bf16*)carve((size_t)SS * DM * 2);       // bf16(query_table)
    __bf16* XKB  = (__bf16*)carve(NX * 2);                    // bf16(x_key) [B*L, D]
    __bf16* E0B  = (__bf16*)carve(NX * 2);                    // bf16(a0/b0) staging
    __bf16* AET  = (__bf16*)carve(NX * 2);                    // a_emb^T  [b, D, L]
    __bf16* BET  = (__bf16*)carve(NX * 2);                    // b_emb^T  [b, D, L]
    __bf16* FWA  = (__bf16*)carve((size_t)BS * SP * LQ * 2);  // fw_a [b, SP, L]
    __bf16* FWB  = (__bf16*)carve((size_t)BS * SP * LQ * 2);
    __bf16* BWA  = (__bf16*)carve((size_t)BS * LQ * SS * 2);  // bw_a [b, L, SS]
    __bf16* BWB  = (__bf16*)carve((size_t)BS * LQ * SS * 2);
    __bf16* CTA  = (__bf16*)carve((size_t)BS * DM * SS * 2);  // class_a^T [b, D, SS]
    __bf16* CTB  = (__bf16*)carve((size_t)BS * DM * SS * 2);
    float*  F1   = (float*)carve(NX * 4);                     // x_key/a0/b0/sel f32
    float*  F2   = (float*)carve(NX * 4);                     // a1/b1/z/outB f32
    float*  F3   = (float*)carve(NX * 4);                     // outA f32
    float*  CF   = (float*)carve((size_t)BS * SP * DM * 4);   // class f32 [b, SP, D]
    (void)ws_size; (void)n_in; (void)in_sizes; (void)out_size;

    // --- stage 0: bf16 conversions ---
    launch_cvt(x,   XB,   NX, stream);
    launch_cvt(Wk,  WKB,  NW, stream);
    launch_cvt(Wa,  WAB,  NW, stream);
    launch_cvt(Wa1, WA1B, NW, stream);
    launch_cvt(Wb,  WBB,  NW, stream);
    launch_cvt(Wb1, WB1B, NW, stream);
    launch_cvt(Wsl, WSB,  NW, stream);
    launch_cvt(qtab, QTB, (size_t)SS * DM, stream);

    const int BL = BS * LQ;  // 16384

    // --- x_key = x @ Wk^T + bk ---
    launch_gemm(XB, WKB, bk, F1, BL, DM, DM, DM, DM, DM, 0, 0, 0, 1, nullptr, 0, 0, 1.0f, stream);
    launch_cvt(F1, XKB, NX, stream);

    // --- a branch: a0 = x@Wa^T+ba ; a1 = a0@Wa1^T+ba1 ; a_emb^T = T(a0*sig(a1)) ---
    launch_gemm(XB, WAB, ba, F1, BL, DM, DM, DM, DM, DM, 0, 0, 0, 1, nullptr, 0, 0, 1.0f, stream);
    launch_cvt(F1, E0B, NX, stream);
    launch_gemm(E0B, WA1B, ba1, F2, BL, DM, DM, DM, DM, DM, 0, 0, 0, 1, nullptr, 0, 0, 1.0f, stream);
    {
        dim3 g(DM / 32, LQ / 32, BS), blk(256);
        hipLaunchKernelGGL(gate_transpose, g, blk, 0, stream, F1, F2, AET);
    }

    // --- b branch ---
    launch_gemm(XB, WBB, bb, F1, BL, DM, DM, DM, DM, DM, 0, 0, 0, 1, nullptr, 0, 0, 1.0f, stream);
    launch_cvt(F1, E0B, NX, stream);
    launch_gemm(E0B, WB1B, bb1, F2, BL, DM, DM, DM, DM, DM, 0, 0, 0, 1, nullptr, 0, 0, 1.0f, stream);
    {
        dim3 g(DM / 32, LQ / 32, BS), blk(256);
        hipLaunchKernelGGL(gate_transpose, g, blk, 0, stream, F1, F2, BET);
    }

    // --- selector logits (sigmoid fused in final combine) ---
    launch_gemm(XB, WSB, bsel, F1, BL, DM, DM, DM, DM, DM, 0, 0, 0, 1, nullptr, 0, 0, 1.0f, stream);

    // --- z[b,s,l] = query_table[nidx[b,s]] . x_key[b,l] / 32  (batched, row-gathered A) ---
    launch_gemm(QTB, XKB, nullptr, F2,
                SP, LQ, DM, DM, DM, LQ,
                0, (long long)LQ * DM, (long long)SP * LQ, BS,
                nidx, SS, SS, 1.0f / 32.0f, stream);

    // --- forward / backward normalizations ---
    {
        dim3 g(SP, BS), blk(256);
        hipLaunchKernelGGL(fw_norm, g, blk, 0, stream, F2, mask, FWA, FWB);
    }
    {
        dim3 g(LQ, BS), blk(256);
        hipLaunchKernelGGL(bw_norm, g, blk, 0, stream, F2, BWA, BWB);
    }

    // --- class_a[b,s,d] = fw_a . a_emb ; +bias_exp ; transpose to [b,d,s] ---
    launch_gemm(FWA, AET, nullptr, CF,
                SP, DM, LQ, LQ, LQ, DM,
                (long long)SP * LQ, (long long)DM * LQ, (long long)SP * DM, BS,
                nullptr, 0, 0, 1.0f, stream);
    {
        dim3 g(SS / 32, DM / 32, BS), blk(256);
        hipLaunchKernelGGL(bias_transpose, g, blk, 0, stream, CF, btab, nidx, CTA);
    }
    launch_gemm(FWB, BET, nullptr, CF,
                SP, DM, LQ, LQ, LQ, DM,
                (long long)SP * LQ, (long long)DM * LQ, (long long)SP * DM, BS,
                nullptr, 0, 0, 1.0f, stream);
    {
        dim3 g(SS / 32, DM / 32, BS), blk(256);
        hipLaunchKernelGGL(bias_transpose, g, blk, 0, stream, CF, btab, nidx, CTB);
    }

    // --- outA[b,l,d] = bw_a . class_a^T ; outB similarly ---
    launch_gemm(BWA, CTA, nullptr, F3,
                LQ, DM, SS, SS, SS, DM,
                (long long)LQ * SS, (long long)DM * SS, (long long)LQ * DM, BS,
                nullptr, 0, 0, 1.0f, stream);
    launch_gemm(BWB, CTB, nullptr, F2,
                LQ, DM, SS, SS, SS, DM,
                (long long)LQ * SS, (long long)DM * SS, (long long)LQ * DM, BS,
                nullptr, 0, 0, 1.0f, stream);

    // --- out = 0.2*(sig(sel)*outA + (1-sig(sel))*outB) ---
    {
        dim3 g((unsigned)((NX + 255) / 256)), blk(256);
        hipLaunchKernelGGL(final_combine, g, blk, 0, stream, F1, F3, F2, out, NX);
    }
}
